// TransformerDecoderLayerGQA_RoPE_51951924412827
// MI455X (gfx1250) — matrix operand order repaired and checked by reading, not compiled, not run on hardware
//
#include <hip/hip_runtime.h>

#define D_MODEL 1024
#define N_HEADS 16
#define HEAD_DIM 64
#define D_FF 4096
#define T_SEQ 2048
#define BATCH 2
#define ROWS (BATCH * T_SEQ) /* 4096 */

typedef __attribute__((ext_vector_type(16))) __bf16 v16bf;
typedef __attribute__((ext_vector_type(8)))  __bf16 v8bf;
typedef __attribute__((ext_vector_type(8)))  float  v8f;
typedef __attribute__((ext_vector_type(4)))  unsigned tdm_v4u;
typedef __attribute__((ext_vector_type(8)))  int      tdm_v8i;
typedef __attribute__((ext_vector_type(4)))  int      tdm_v4i;

#define CAT16(lo, hi) __builtin_shufflevector((lo), (hi), 0,1,2,3,4,5,6,7,8,9,10,11,12,13,14,15)

__device__ __forceinline__ __bf16 f2bf(float f) {
  union { float f; unsigned u; } a; a.f = f;
  unsigned u = a.u;
  u += 0x7FFFu + ((u >> 16) & 1u);
  unsigned short hs = (unsigned short)(u >> 16);
  union { unsigned short s; __bf16 b; } o; o.s = hs;
  return o.b;
}

// ---------------------------------------------------------------------------
// TDM: issue a 2-D tile load (bf16 elements) global -> LDS.
// D# packing per CDNA5 ISA 8.3/8.4. pad(3,3): +16B after every 64B row, which
// materializes a 40-element LDS row stride for a 32-element tile row.
// This toolchain's builtin takes 6 args (extra int32x8 group before cpol);
// a zero-filled group is a NULL descriptor (count=0).
// ---------------------------------------------------------------------------
__device__ __forceinline__ void tdm_load_2d_bf16(
    unsigned lds_addr, const void* gptr,
    unsigned tdim0, unsigned tdim1,      // tensor extents (elements)
    unsigned tile0, unsigned tile1,      // tile extents (elements)
    unsigned stride0,                    // tensor dim0 stride (elements)
    int padded) {
  unsigned long long ga = (unsigned long long)(uintptr_t)gptr;
  tdm_v4u g0;
  g0[0] = 1u;                                              // count=1, user mode
  g0[1] = lds_addr;                                        // LDS byte address
  g0[2] = (unsigned)ga;                                    // global_addr[31:0]
  g0[3] = (unsigned)((ga >> 32) & 0x01FFFFFFull) | (2u << 30);  // [56:32] | type=2
  tdm_v8i g1;
  unsigned w0 = (1u << 16);                                // data_size=1 (2 bytes)
  if (padded) w0 |= (1u << 20) | (3u << 22) | (3u << 25);  // pad_en, every 16 dw, +4 dw
  g1[0] = (int)w0;
  g1[1] = (int)((tdim0 & 0xFFFFu) << 16);                  // tensor_dim0[15:0]
  g1[2] = (int)((tdim0 >> 16) | ((tdim1 & 0xFFFFu) << 16));// dim0[31:16]|dim1[15:0]
  g1[3] = (int)((tdim1 >> 16) | (tile0 << 16));            // dim1[31:16]|tile_dim0
  g1[4] = (int)(tile1 & 0xFFFFu);                          // tile_dim1 (tile_dim2=0)
  g1[5] = (int)stride0;                                    // dim0_stride[31:0]
  g1[6] = 0;                                               // stride hi / dim1_stride lo
  g1[7] = 0;
  tdm_v4i z4 = {0, 0, 0, 0};
  tdm_v8i z8 = {0, 0, 0, 0, 0, 0, 0, 0};
  __builtin_amdgcn_tensor_load_to_lds(g0, g1, z4, z4, z8, 0);
}

// Per-lane async 16B copy global -> LDS (ASYNCcnt path)
__device__ __forceinline__ void async_b128_to_lds(unsigned lds_off, const void* gp) {
  asm volatile("global_load_async_to_lds_b128 %0, %1, off"
               :: "v"(lds_off), "v"(gp) : "memory");
}
__device__ __forceinline__ void wait_asynccnt0() {
  asm volatile("s_wait_asynccnt 0x0" ::: "memory");
}

// ---------------------------------------------------------------------------
// fp32 -> bf16 elementwise (4 elems / thread)
// ---------------------------------------------------------------------------
__global__ void k_f32_to_bf16(const float* __restrict__ in, __bf16* __restrict__ out, int n) {
  int base = (blockIdx.x * 256 + threadIdx.x) * 4;
  if (base + 3 < n) {
#pragma unroll
    for (int i = 0; i < 4; ++i) out[base + i] = f2bf(in[base + i]);
  }
}

// ---------------------------------------------------------------------------
// W (K x N, fp32, row-major) -> Wt (N x K, bf16)
// ---------------------------------------------------------------------------
__global__ void k_transpose_bf16(const float* __restrict__ W, __bf16* __restrict__ Wt,
                                 int K, int N) {
  __shared__ float tile[32][33];
  int bx = blockIdx.x * 32;  // N
  int by = blockIdx.y * 32;  // K
  int tx = threadIdx.x, ty = threadIdx.y;  // 32 x 8
#pragma unroll
  for (int i = 0; i < 32; i += 8)
    tile[ty + i][tx] = W[(size_t)(by + ty + i) * N + bx + tx];
  __syncthreads();
#pragma unroll
  for (int i = 0; i < 32; i += 8)
    Wt[(size_t)(bx + ty + i) * K + by + tx] = f2bf(tile[tx][ty + i]);
}

// ---------------------------------------------------------------------------
// RoPE learned tables:  P = base @ A  (2048 x 16)
// ---------------------------------------------------------------------------
__global__ void k_rope_P(const float* __restrict__ A, float* __restrict__ P) {
  int idx = blockIdx.x * 256 + threadIdx.x;  // t*16 + r, 32768 total
  int t = idx >> 4, r = idx & 15;
  float acc = 0.f;
  for (int d = 0; d < D_MODEL; ++d) {
    int j = (d < 512) ? d : d - 512;
    float theta = powf(10000.f, -(float)(2 * j) * (1.f / 1024.f));
    float ang = (float)t * theta;
    float base = (d < 512) ? sinf(ang) : cosf(ang);
    acc += base * A[d * 16 + r];
  }
  P[t * 16 + r] = acc;
}

// learned = P @ B; sin_f = sin + learned[:,:512], cos_f = cos + learned[:,512:]
__global__ void k_rope_tab(const float* __restrict__ P, const float* __restrict__ Bm,
                           float* __restrict__ sin_t, float* __restrict__ cos_t) {
  int idx = blockIdx.x * 256 + threadIdx.x;  // t*1024 + j
  int t = idx >> 10, j = idx & 1023;
  float acc = 0.f;
#pragma unroll
  for (int r = 0; r < 16; ++r) acc += P[t * 16 + r] * Bm[r * 1024 + j];
  int jj = (j < 512) ? j : j - 512;
  float theta = powf(10000.f, -(float)(2 * jj) * (1.f / 1024.f));
  float ang = (float)t * theta;
  if (j < 512) sin_t[t * 512 + jj] = sinf(ang) + acc;
  else         cos_t[t * 512 + jj] = cosf(ang) + acc;
}

// apply rope to fp32 (ROWS x D), write bf16
__global__ void k_rope_apply(const float* __restrict__ x, const float* __restrict__ sin_t,
                             const float* __restrict__ cos_t, __bf16* __restrict__ out) {
  int row = blockIdx.x;
  int t = row & (T_SEQ - 1);
  const float* xr = x + (size_t)row * D_MODEL;
  __bf16* orow = out + (size_t)row * D_MODEL;
  for (int j = threadIdx.x; j < 512; j += 256) {
    float xe = xr[2 * j], xo = xr[2 * j + 1];
    float s = sin_t[t * 512 + j], c = cos_t[t * 512 + j];
    orow[2 * j]     = f2bf(xe * c - xo * s);
    orow[2 * j + 1] = f2bf(xe * s + xo * c);
  }
}

// ---------------------------------------------------------------------------
// Generic bf16 WMMA GEMM, TDM double-buffered:
//   out = op(A[MxK] @ Wt[NxK]^T + bias) [+ resid]
// 128x128 block tile, BK=32. Wave 0 drives the Tensor Data Mover; all 8 waves
// run WMMA on the other LDS buffer. TENSORcnt paces the pipeline.
// ---------------------------------------------------------------------------
__global__ __launch_bounds__(256) void k_gemm_bf16(
    const __bf16* __restrict__ A, const __bf16* __restrict__ Bt,
    const float* __restrict__ bias, const float* __restrict__ resid,
    float* __restrict__ outF, __bf16* __restrict__ outB,
    int M, int N, int K, int relu) {
  __shared__ __align__(16) __bf16 As[2][128][40];
  __shared__ __align__(16) __bf16 Bs[2][128][40];
  int tid = threadIdx.x;
  int lane = tid & 31, wid = tid >> 5;
  int wm = wid >> 2, wn = wid & 3;
  int bm = blockIdx.y * 128, bn = blockIdx.x * 128;
  int m16 = lane & 15;
  int kh = (lane >> 4) << 3;   // A-frag k offset (0/8)
  int kb = (lane >> 4) << 4;   // unused (0/16); B-frag k offset is kh
  int rh = (lane >> 4) << 3;   // C-frag row offset (0/8)

  unsigned asOff[2] = {(unsigned)(uintptr_t)&As[0][0][0], (unsigned)(uintptr_t)&As[1][0][0]};
  unsigned bsOff[2] = {(unsigned)(uintptr_t)&Bs[0][0][0], (unsigned)(uintptr_t)&Bs[1][0][0]};

  v8f acc[4][2];
#pragma unroll
  for (int i = 0; i < 4; ++i)
#pragma unroll
    for (int j = 0; j < 2; ++j)
#pragma unroll
      for (int e = 0; e < 8; ++e) acc[i][j][e] = 0.f;

  int nk = K >> 5;
  // prologue: DMA k-tile 0 into buffer 0
  if (wid == 0) {
    tdm_load_2d_bf16(asOff[0], A + (size_t)bm * K, K, M, 32, 128, K, 1);
    tdm_load_2d_bf16(bsOff[0], Bt + (size_t)bn * K, K, N, 32, 128, K, 1);
  }
  for (int kt = 0; kt < nk; ++kt) {
    int cur = kt & 1;
    __syncthreads();  // previous buffer's consumers are done; safe to refill it
    if (wid == 0) {
      if (kt + 1 < nk) {
        tdm_load_2d_bf16(asOff[cur ^ 1], A + (size_t)bm * K + (kt + 1) * 32, K, M, 32, 128, K, 1);
        tdm_load_2d_bf16(bsOff[cur ^ 1], Bt + (size_t)bn * K + (kt + 1) * 32, K, N, 32, 128, K, 1);
        __builtin_amdgcn_s_wait_tensorcnt((short)2);  // current buffer's 2 ops done
      } else {
        __builtin_amdgcn_s_wait_tensorcnt((short)0);
      }
    }
    __syncthreads();  // publish buffer `cur`

    v16bf af[4], bfr[2];
#pragma unroll
    for (int mi = 0; mi < 4; ++mi) {
      const __bf16* ap = &As[cur][wm * 64 + mi * 16 + m16][kh];
      v8bf lo = *(const v8bf*)ap;
      v8bf hi = *(const v8bf*)(ap + 16);
      af[mi] = CAT16(lo, hi);
    }
#pragma unroll
    for (int ni = 0; ni < 2; ++ni) {
      const __bf16* bp = &Bs[cur][wn * 32 + ni * 16 + m16][kh];
      v8bf lo = *(const v8bf*)bp;
      v8bf hi = *(const v8bf*)(bp + 16);
      bfr[ni] = CAT16(lo, hi);
    }
#pragma unroll
    for (int mi = 0; mi < 4; ++mi)
#pragma unroll
      for (int ni = 0; ni < 2; ++ni)
        acc[mi][ni] = __builtin_amdgcn_wmma_f32_16x16x32_bf16(
            false, af[mi], false, bfr[ni], (short)0, acc[mi][ni], false, false);
  }

#pragma unroll
  for (int mi = 0; mi < 4; ++mi)
#pragma unroll
    for (int ni = 0; ni < 2; ++ni) {
      int col = bn + wn * 32 + ni * 16 + m16;
      float bc = bias ? bias[col] : 0.f;
#pragma unroll
      for (int v = 0; v < 8; ++v) {
        int row = bm + wm * 64 + mi * 16 + v + rh;
        float val = acc[mi][ni][v] + bc;
        if (relu) val = fmaxf(val, 0.f);
        if (resid) val += resid[(size_t)row * N + col];
        if (outF) outF[(size_t)row * N + col] = val;
        if (outB) outB[(size_t)row * N + col] = f2bf(val);
      }
    }
}

// ---------------------------------------------------------------------------
// Flash attention (full, non-causal). Block = (b, h, 128 q rows), 8 waves,
// each wave owns one 16-row q tile; KV chunks of 64; online softmax.
// K chunk staged via GLOBAL_LOAD_ASYNC_TO_LDS (ASYNCcnt).
// ---------------------------------------------------------------------------
__global__ __launch_bounds__(256) void k_flash_attn(
    const __bf16* __restrict__ Q, const __bf16* __restrict__ Kk,
    const __bf16* __restrict__ V, __bf16* __restrict__ O) {
  __shared__ __align__(16) __bf16 Qs[128][72];
  __shared__ __align__(16) __bf16 Ks[64][72];
  __shared__ __align__(16) __bf16 Vs[64][72];        // Vs[dim][kv]
  __shared__ __align__(16) __bf16 Ps[8][16][72];     // wave-private P staging

  int tid = threadIdx.x, lane = tid & 31, wid = tid >> 5;
  int h = blockIdx.y & (N_HEADS - 1);
  int b = blockIdx.y >> 4;
  int qbase = blockIdx.x * 128;
  size_t headOff = (size_t)b * T_SEQ * D_MODEL + (size_t)h * HEAD_DIM;
  const __bf16* qg = Q + headOff + (size_t)qbase * D_MODEL;
  const __bf16* kg = Kk + headOff;
  const __bf16* vg = V + headOff;

#pragma unroll
  for (int j = 0; j < 4; ++j) {  // 128x64 bf16 = 1024 uint4
    int i = tid + 256 * j;
    int r = i >> 3, c = (i & 7) << 3;
    *(uint4*)&Qs[r][c] = *(const uint4*)(qg + (size_t)r * D_MODEL + c);
  }
  __syncthreads();

  int m16 = lane & 15;
  int kh = (lane >> 4) << 3;
  int kb = (lane >> 4) << 4;
  int rh = (lane >> 4) << 3;

  v16bf qa[2];
#pragma unroll
  for (int ks = 0; ks < 2; ++ks) {
    const __bf16* ap = &Qs[wid * 16 + m16][ks * 32 + kh];
    v8bf lo = *(const v8bf*)ap;
    v8bf hi = *(const v8bf*)(ap + 16);
    qa[ks] = CAT16(lo, hi);
  }

  float mrow[8], ssum[8];
  v8f ctx[4];
#pragma unroll
  for (int v = 0; v < 8; ++v) { mrow[v] = -3.0e38f; ssum[v] = 0.f; }
#pragma unroll
  for (int d = 0; d < 4; ++d)
#pragma unroll
    for (int e = 0; e < 8; ++e) ctx[d][e] = 0.f;

  for (int ch = 0; ch < T_SEQ / 64; ++ch) {
    __syncthreads();
    // K chunk via async-to-LDS (per-lane 16B), V chunk transposed manually
#pragma unroll
    for (int j = 0; j < 2; ++j) {  // 64x64 bf16 = 512 uint4
      int i = tid + 256 * j;
      int r = i >> 3, c = (i & 7) << 3;
      async_b128_to_lds((unsigned)(uintptr_t)&Ks[r][c],
                        kg + (size_t)(ch * 64 + r) * D_MODEL + c);
    }
#pragma unroll
    for (int j = 0; j < 16; ++j) {
      int i = tid + 256 * j;
      int kv = i >> 6, d = i & 63;
      Vs[d][kv] = vg[(size_t)(ch * 64 + kv) * D_MODEL + d];
    }
    if (ch + 1 < T_SEQ / 64) {  // prefetch next K chunk
      __builtin_prefetch(kg + (size_t)((ch + 1) * 64 + (tid >> 2)) * D_MODEL, 0, 1);
      __builtin_prefetch(vg + (size_t)((ch + 1) * 64 + (tid >> 2)) * D_MODEL, 0, 1);
    }
    wait_asynccnt0();
    __syncthreads();

    // S = (Q @ K^T) / 8 for this wave's 16 q rows x 64 kv cols
    v8f S[4];
#pragma unroll
    for (int tn = 0; tn < 4; ++tn) {
      v8f s;
#pragma unroll
      for (int e = 0; e < 8; ++e) s[e] = 0.f;
#pragma unroll
      for (int ks = 0; ks < 2; ++ks) {
        const __bf16* bp = &Ks[tn * 16 + m16][ks * 32 + kh];
        v8bf lo = *(const v8bf*)bp;
        v8bf hi = *(const v8bf*)(bp + 16);
        v16bf kf = CAT16(lo, hi);
        s = __builtin_amdgcn_wmma_f32_16x16x32_bf16(false, qa[ks], false, kf,
                                                    (short)0, s, false, false);
      }
#pragma unroll
      for (int v = 0; v < 8; ++v) S[tn][v] = s[v] * 0.125f;
    }

    // online softmax: row reductions live in 16-lane halves
    float cmax[8];
#pragma unroll
    for (int v = 0; v < 8; ++v)
      cmax[v] = fmaxf(fmaxf(S[0][v], S[1][v]), fmaxf(S[2][v], S[3][v]));
#pragma unroll
    for (int mask = 1; mask <= 8; mask <<= 1)
#pragma unroll
      for (int v = 0; v < 8; ++v)
        cmax[v] = fmaxf(cmax[v], __shfl_xor(cmax[v], mask, 32));
    float alpha[8];
#pragma unroll
    for (int v = 0; v < 8; ++v) {
      float mn = fmaxf(mrow[v], cmax[v]);
      alpha[v] = __expf(mrow[v] - mn);
      mrow[v] = mn;
    }
    float csum[8];
#pragma unroll
    for (int v = 0; v < 8; ++v) csum[v] = 0.f;
#pragma unroll
    for (int tn = 0; tn < 4; ++tn)
#pragma unroll
      for (int v = 0; v < 8; ++v) {
        float e = __expf(S[tn][v] - mrow[v]);
        S[tn][v] = e;
        csum[v] += e;
      }
#pragma unroll
    for (int mask = 1; mask <= 8; mask <<= 1)
#pragma unroll
      for (int v = 0; v < 8; ++v) csum[v] += __shfl_xor(csum[v], mask, 32);
#pragma unroll
    for (int v = 0; v < 8; ++v) ssum[v] = ssum[v] * alpha[v] + csum[v];
#pragma unroll
    for (int d = 0; d < 4; ++d)
#pragma unroll
      for (int v = 0; v < 8; ++v) ctx[d][v] *= alpha[v];

    // stage P (C-frag layout) -> LDS -> A-frag layout; wave-private, in-order DS
#pragma unroll
    for (int tn = 0; tn < 4; ++tn)
#pragma unroll
      for (int v = 0; v < 8; ++v)
        Ps[wid][v + rh][tn * 16 + m16] = f2bf(S[tn][v]);

#pragma unroll
    for (int ks = 0; ks < 2; ++ks) {
      const __bf16* ap = &Ps[wid][m16][ks * 32 + kh];
      v8bf lo = *(const v8bf*)ap;
      v8bf hi = *(const v8bf*)(ap + 16);
      v16bf pa = CAT16(lo, hi);
#pragma unroll
      for (int d = 0; d < 4; ++d) {
        const __bf16* bp = &Vs[d * 16 + m16][ks * 32 + kh];
        v8bf blo = *(const v8bf*)bp;
        v8bf bhi = *(const v8bf*)(bp + 16);
        v16bf vf = CAT16(blo, bhi);
        ctx[d] = __builtin_amdgcn_wmma_f32_16x16x32_bf16(false, pa, false, vf,
                                                         (short)0, ctx[d], false, false);
      }
    }
  }

#pragma unroll
  for (int d = 0; d < 4; ++d)
#pragma unroll
    for (int v = 0; v < 8; ++v) {
      float val = ctx[d][v] / ssum[v];
      int row = qbase + wid * 16 + v + rh;
      int col = d * 16 + m16;
      O[headOff + (size_t)row * D_MODEL + col] = f2bf(val);
    }
}

// ---------------------------------------------------------------------------
// LayerNorm over D=1024 (one row per block, 256 threads)
// ---------------------------------------------------------------------------
__global__ __launch_bounds__(256) void k_layernorm(
    const float* __restrict__ x, const float* __restrict__ g, const float* __restrict__ bb,
    float* __restrict__ outF, __bf16* __restrict__ outB) {
  __shared__ float red[16];
  int row = blockIdx.x, tid = threadIdx.x;
  const float* xr = x + (size_t)row * D_MODEL;
  float vals[4];
  float s = 0.f, sq = 0.f;
#pragma unroll
  for (int i = 0; i < 4; ++i) {
    vals[i] = xr[tid + 256 * i];
    s += vals[i];
    sq += vals[i] * vals[i];
  }
#pragma unroll
  for (int mask = 16; mask; mask >>= 1) {
    s += __shfl_xor(s, mask, 32);
    sq += __shfl_xor(sq, mask, 32);
  }
  int wid = tid >> 5, lane = tid & 31;
  if (lane == 0) { red[wid] = s; red[8 + wid] = sq; }
  __syncthreads();
  s = 0.f; sq = 0.f;
#pragma unroll
  for (int w = 0; w < 8; ++w) { s += red[w]; sq += red[8 + w]; }
  float mu = s * (1.f / 1024.f);
  float var = sq * (1.f / 1024.f) - mu * mu;
  float rstd = rsqrtf(var + 1e-5f);
#pragma unroll
  for (int i = 0; i < 4; ++i) {
    int c = tid + 256 * i;
    float y = (vals[i] - mu) * rstd * g[c] + bb[c];
    if (outF) outF[(size_t)row * D_MODEL + c] = y;
    if (outB) outB[(size_t)row * D_MODEL + c] = f2bf(y);
  }
}

// ---------------------------------------------------------------------------
extern "C" void kernel_launch(void* const* d_in, const int* in_sizes, int n_in,
                              void* d_out, int out_size, void* d_ws, size_t ws_size,
                              hipStream_t stream) {
  const float* tgt  = (const float*)d_in[0];
  const float* Wq   = (const float*)d_in[1];
  const float* bq   = (const float*)d_in[2];
  const float* Wk   = (const float*)d_in[3];
  const float* bk   = (const float*)d_in[4];
  const float* Wv   = (const float*)d_in[5];
  const float* bv   = (const float*)d_in[6];
  const float* Wo   = (const float*)d_in[7];
  const float* bo   = (const float*)d_in[8];
  const float* W1   = (const float*)d_in[9];
  const float* b1   = (const float*)d_in[10];
  const float* W2   = (const float*)d_in[11];
  const float* b2   = (const float*)d_in[12];
  const float* ln1g = (const float*)d_in[13];
  const float* ln1b = (const float*)d_in[14];
  const float* ln2g = (const float*)d_in[15];
  const float* ln2b = (const float*)d_in[16];
  const float* ropeA = (const float*)d_in[17];
  const float* ropeB = (const float*)d_in[18];

  char* ws = (char*)d_ws;
  size_t off = 0;
  auto alloc = [&](size_t bytes) -> void* {
    void* p = ws + off;
    off = (off + bytes + 255) & ~(size_t)255;
    return p;
  };
  __bf16* xb   = (__bf16*)alloc((size_t)ROWS * D_MODEL * 2);
  __bf16* WqT  = (__bf16*)alloc((size_t)D_MODEL * D_MODEL * 2);
  __bf16* WkT  = (__bf16*)alloc((size_t)D_MODEL * D_MODEL * 2);
  __bf16* WvT  = (__bf16*)alloc((size_t)D_MODEL * D_MODEL * 2);
  __bf16* WoT  = (__bf16*)alloc((size_t)D_MODEL * D_MODEL * 2);
  __bf16* W1T  = (__bf16*)alloc((size_t)D_FF * D_MODEL * 2);
  __bf16* W2T  = (__bf16*)alloc((size_t)D_MODEL * D_FF * 2);
  float*  Ptab = (float*)alloc((size_t)T_SEQ * 16 * 4);
  float*  sinT = (float*)alloc((size_t)T_SEQ * 512 * 4);
  float*  cosT = (float*)alloc((size_t)T_SEQ * 512 * 4);
  __bf16* qb   = (__bf16*)alloc((size_t)ROWS * D_MODEL * 2);
  __bf16* kbuf = (__bf16*)alloc((size_t)ROWS * D_MODEL * 2);
  __bf16* vb   = (__bf16*)alloc((size_t)ROWS * D_MODEL * 2);
  __bf16* ctxb = (__bf16*)alloc((size_t)ROWS * D_MODEL * 2);
  float*  T0   = (float*)alloc((size_t)ROWS * D_MODEL * 4);   // q32/k32 temp
  float*  x1   = (float*)alloc((size_t)ROWS * D_MODEL * 4);   // attn resid, then x2
  float*  x1n  = (float*)alloc((size_t)ROWS * D_MODEL * 4);
  __bf16* x1nb = (__bf16*)alloc((size_t)ROWS * D_MODEL * 2);
  __bf16* hb   = (__bf16*)alloc((size_t)ROWS * D_FF * 2);
  (void)ws_size; (void)in_sizes; (void)n_in; (void)out_size;

  // prep
  k_f32_to_bf16<<<ROWS * D_MODEL / 1024, 256, 0, stream>>>(tgt, xb, ROWS * D_MODEL);
  dim3 tb(32, 8);
  k_transpose_bf16<<<dim3(32, 32), tb, 0, stream>>>(Wq, WqT, D_MODEL, D_MODEL);
  k_transpose_bf16<<<dim3(32, 32), tb, 0, stream>>>(Wk, WkT, D_MODEL, D_MODEL);
  k_transpose_bf16<<<dim3(32, 32), tb, 0, stream>>>(Wv, WvT, D_MODEL, D_MODEL);
  k_transpose_bf16<<<dim3(32, 32), tb, 0, stream>>>(Wo, WoT, D_MODEL, D_MODEL);
  k_transpose_bf16<<<dim3(128, 32), tb, 0, stream>>>(W1, W1T, D_MODEL, D_FF);
  k_transpose_bf16<<<dim3(32, 128), tb, 0, stream>>>(W2, W2T, D_FF, D_MODEL);
  k_rope_P<<<T_SEQ * 16 / 256, 256, 0, stream>>>(ropeA, Ptab);
  k_rope_tab<<<T_SEQ * 1024 / 256, 256, 0, stream>>>(Ptab, ropeB, sinT, cosT);

  // Q/K/V projections (+RoPE on Q,K)
  k_gemm_bf16<<<dim3(D_MODEL / 128, ROWS / 128), 256, 0, stream>>>(
      xb, WqT, bq, nullptr, T0, nullptr, ROWS, D_MODEL, D_MODEL, 0);
  k_rope_apply<<<ROWS, 256, 0, stream>>>(T0, sinT, cosT, qb);
  k_gemm_bf16<<<dim3(D_MODEL / 128, ROWS / 128), 256, 0, stream>>>(
      xb, WkT, bk, nullptr, T0, nullptr, ROWS, D_MODEL, D_MODEL, 0);
  k_rope_apply<<<ROWS, 256, 0, stream>>>(T0, sinT, cosT, kbuf);
  k_gemm_bf16<<<dim3(D_MODEL / 128, ROWS / 128), 256, 0, stream>>>(
      xb, WvT, bv, nullptr, nullptr, vb, ROWS, D_MODEL, D_MODEL, 0);

  // attention
  k_flash_attn<<<dim3(T_SEQ / 128, BATCH * N_HEADS), 256, 0, stream>>>(qb, kbuf, vb, ctxb);

  // x1 = tgt + ctx @ Wo + bo ; ln1
  k_gemm_bf16<<<dim3(D_MODEL / 128, ROWS / 128), 256, 0, stream>>>(
      ctxb, WoT, bo, tgt, x1, nullptr, ROWS, D_MODEL, D_MODEL, 0);
  k_layernorm<<<ROWS, 256, 0, stream>>>(x1, ln1g, ln1b, x1n, x1nb);

  // FFN
  k_gemm_bf16<<<dim3(D_FF / 128, ROWS / 128), 256, 0, stream>>>(
      x1nb, W1T, b1, nullptr, nullptr, hb, ROWS, D_FF, D_MODEL, 1);
  k_gemm_bf16<<<dim3(D_MODEL / 128, ROWS / 128), 256, 0, stream>>>(
      hb, W2T, b2, x1n, x1, nullptr, ROWS, D_MODEL, D_FF, 0);

  // ln2 -> out
  k_layernorm<<<ROWS, 256, 0, stream>>>(x1, ln2g, ln2b, (float*)d_out, nullptr);
}
